// LogicGatedSNN_29867202576804
// MI455X (gfx1250) — matrix operand order repaired and checked by reading, not compiled, not run on hardware
//
#include <hip/hip_runtime.h>
#include <cstdint>
#include <cstddef>

typedef _Float16 v16h __attribute__((ext_vector_type(16)));
typedef float    v8f  __attribute__((ext_vector_type(8)));
typedef float    f32x4 __attribute__((ext_vector_type(4)));

#define IN_F  8192
#define OUT_F 8192
#define THRESH 50.0f

// ---------------------------------------------------------------------------
// Kernel 1: binarized matvec via WMMA + LIF update.
// Block = 256 threads (8 waves). Each block owns 16 output rows; the 8 waves
// split the K=8192 reduction (interleaved 32-wide chunks -> contiguous
// streaming). One v_wmma_f32_16x16x32_f16 per 16x32 tile computes BOTH
// sum(w*x) (B col 0 = x) and sum(w) (B col 1 = ones) exactly in f32.
// ---------------------------------------------------------------------------
__global__ __launch_bounds__(256) void snn_matvec_wmma(
    const float* __restrict__ x,
    const float* __restrict__ states,
    const float* __restrict__ mp,
    const float* __restrict__ thr,
    float* __restrict__ out_spk,
    float* __restrict__ out_v,
    float* __restrict__ out_thr)
{
  __shared__ float s_cur[16];
  __shared__ float s_conn[16];
  const int tid = threadIdx.x;
  if (tid < 16) { s_cur[tid] = 0.0f; s_conn[tid] = 0.0f; }
  __syncthreads();

  const int lane = tid & 31;
  const int wave = tid >> 5;        // 0..7, K-splitting
  const int r15  = lane & 15;       // row within tile (A) / column (B,D)
  const int g    = lane >> 4;       // half-wave group
  const int rowBase = blockIdx.x << 4;
  const int row = rowBase + r15;

  const float* srow = states + (size_t)row * IN_F;

  v8f c = {};
  for (int it = 0; it < 32; ++it) {
    const int k0 = it * 256 + wave * 32;   // 8 waves cover 256 contiguous cols/iter

    // ---- A tile: w = (states > 50) as f16 0/1 ------------------------------
    // lane g=0: K {k0..k0+7, k0+16..k0+23}; g=1: K {k0+8..k0+15, k0+24..k0+31}
    const float* sp = srow + k0 + g * 8;
    f32x4 p0 = __builtin_nontemporal_load((const f32x4*)(sp));
    f32x4 p1 = __builtin_nontemporal_load((const f32x4*)(sp + 4));
    f32x4 p2 = __builtin_nontemporal_load((const f32x4*)(sp + 16));
    f32x4 p3 = __builtin_nontemporal_load((const f32x4*)(sp + 20));
    v16h a;
#pragma unroll
    for (int j = 0; j < 4; ++j) {
      a[j]      = (_Float16)((p0[j] > THRESH) ? 1.0f : 0.0f);
      a[4 + j]  = (_Float16)((p1[j] > THRESH) ? 1.0f : 0.0f);
      a[8 + j]  = (_Float16)((p2[j] > THRESH) ? 1.0f : 0.0f);
      a[12 + j] = (_Float16)((p3[j] > THRESH) ? 1.0f : 0.0f);
    }

    // ---- B tile: column 0 = x chunk, column 1 = ones, rest zero ------------
    // lane n holds column n, K = {g*8..g*8+7, 16+g*8..16+g*8+7}
    v16h b = {};
    if (r15 == 0) {
      const float* xp = x + k0 + g * 8;
#pragma unroll
      for (int j = 0; j < 16; ++j) b[j] = (_Float16)xp[(j < 8) ? j : (j + 8)];
    } else if (r15 == 1) {
#pragma unroll
      for (int j = 0; j < 16; ++j) b[j] = (_Float16)1.0f;
    }

    c = __builtin_amdgcn_wmma_f32_16x16x32_f16(
        /*neg_a=*/false, a, /*neg_b=*/false, b,
        /*c_mod=*/(short)0, c, /*reuse_a=*/false, /*reuse_b=*/false);
  }

  // D layout: lane (N=0) 0 -> rows 0..7, lane 16 -> rows 8..15 (current);
  //           lane (N=1) 1 -> rows 0..7, lane 17 -> rows 8..15 (conn_sum).
  if (lane == 0) {
#pragma unroll
    for (int r = 0; r < 8; ++r) atomicAdd(&s_cur[r], c[r]);
  } else if (lane == 16) {
#pragma unroll
    for (int r = 0; r < 8; ++r) atomicAdd(&s_cur[8 + r], c[r]);
  } else if (lane == 1) {
#pragma unroll
    for (int r = 0; r < 8; ++r) atomicAdd(&s_conn[r], c[r]);
  } else if (lane == 17) {
#pragma unroll
    for (int r = 0; r < 8; ++r) atomicAdd(&s_conn[8 + r], c[r]);
  }
  __syncthreads();

  // ---- LIF scalar update for this block's 16 rows --------------------------
  if (tid < 16) {
    const int orow = rowBase + tid;
    const float conn = fmaxf(s_conn[tid], 5.0f);
    const float cur  = s_cur[tid] * (15.0f / sqrtf(conn));
    const float t    = thr[orow];
    const float v    = mp[orow] * 0.85f + cur;
    const float spike = (v >= t) ? 1.0f : 0.0f;
    out_spk[orow] = spike;
    out_v[orow]   = v * (1.0f - spike) * 0.1f;
    const float nt = t + (spike - 0.1f) * 0.1f;
    out_thr[orow] = fminf(fmaxf(nt, 2.0f), 15.0f);
  }
}

// ---------------------------------------------------------------------------
// Kernel 2: eligibility-trace stream (512 MB): one float4 per thread,
// non-temporal so the stream doesn't evict x/spikes from L2.
// ---------------------------------------------------------------------------
__global__ __launch_bounds__(256) void snn_trace(
    const float* __restrict__ x,
    const float* __restrict__ trace,
    const float* __restrict__ spikes,
    float* __restrict__ out_trace)
{
  const size_t idx = (size_t)blockIdx.x * 256 + threadIdx.x;  // float4 index
  const int row = (int)(idx >> 11);        // 2048 float4 per row
  const int c4  = (int)(idx & 2047);
  const float s = spikes[row];
  f32x4 t  = __builtin_nontemporal_load((const f32x4*)trace + idx);
  f32x4 xv = *((const f32x4*)x + c4);      // reused 8192x -> keep cacheable
  f32x4 r;
#pragma unroll
  for (int j = 0; j < 4; ++j) {
    const float v = t[j] * 0.9f + s * xv[j];
    r[j] = fminf(fmaxf(v, 0.0f), 5.0f);
  }
  __builtin_nontemporal_store(r, (f32x4*)out_trace + idx);
}

extern "C" void kernel_launch(void* const* d_in, const int* in_sizes, int n_in,
                              void* d_out, int out_size, void* d_ws, size_t ws_size,
                              hipStream_t stream) {
  const float* x      = (const float*)d_in[0];   // spike_input      [8192]
  const float* states = (const float*)d_in[1];   // synapse_states   [8192,8192]
  const float* mp     = (const float*)d_in[2];   // membrane_pot     [8192]
  const float* thr    = (const float*)d_in[3];   // adaptive_thresh  [8192]
  const float* trace  = (const float*)d_in[4];   // eligibility      [8192,8192]

  float* out       = (float*)d_out;
  float* out_spk   = out;                 // [8192]
  float* out_v     = out + OUT_F;         // [8192]
  float* out_thr   = out + 2 * OUT_F;     // [8192]
  float* out_trace = out + 3 * OUT_F;     // [8192*8192]

  snn_matvec_wmma<<<OUT_F / 16, 256, 0, stream>>>(x, states, mp, thr,
                                                  out_spk, out_v, out_thr);
  const int trace_blocks = (OUT_F * (IN_F / 4)) / 256;  // 65536
  snn_trace<<<trace_blocks, 256, 0, stream>>>(x, trace, out_spk, out_trace);
}
